// DongTaiBaGuaZhen_26079041421869
// MI455X (gfx1250) — matrix operand order repaired and checked by reading, not compiled, not run on hardware
//
#include <hip/hip_runtime.h>

// ---------------------------------------------------------------------------
// MI455X / gfx1250 implementation.
//
// Precision choice (roofline): 137 GFLOP of GEMM vs ~0.6 GB of HBM traffic
// (~26us @ 23.3 TB/s).  fp32 WMMA (16x16x4 = 2048 FLOP/inst) would leave us
// ~8x compute-bound; bf16 WMMA (16x16x32 = 16384 FLOP/inst) with fp32
// accumulators brings GEMM time to the memory floor.  GEMM inputs are rounded
// to bf16; accumulation, means, the 8x8 impedance MLP, head mixing, residual
// and LayerNorm are all fp32.
//
// CDNA5 data-movement: phase-1 stages its shared A-tile through LDS with
// double-buffered GLOBAL_LOAD_ASYNC_TO_LDS_B128 (ASYNCcnt) so the 8 waves of
// a block stop re-fetching the same 16x32 fragment 8x from global.
// ---------------------------------------------------------------------------

typedef __attribute__((ext_vector_type(16))) __bf16 v16bf;
typedef __attribute__((ext_vector_type(8)))  __bf16 v8bf;
typedef __attribute__((ext_vector_type(4)))  __bf16 v4bf;
typedef __attribute__((ext_vector_type(8)))  float  v8f;

#define NB   4       // batch
#define NS   8192    // seq
#define ND   1024    // model dim
#define NHD  128     // head dim
#define NH   8       // heads
#define NP   32
#define NM   (NB*NS) // 32768 rows
#define PI_F 3.14159265358979323846f

// Wave-relative LDS byte offset of a shared-memory pointer (AS3 is 32-bit).
__device__ __forceinline__ unsigned int lds_off(const void* p) {
  return (unsigned int)(size_t)(const __attribute__((address_space(3))) void*)p;
}

__device__ __forceinline__ void async_copy_b128_to_lds(const void* gsrc, void* ldst) {
  asm volatile("global_load_async_to_lds_b128 %0, %1, off"
               :: "v"(lds_off(ldst)), "v"((unsigned long long)(size_t)gsrc)
               : "memory");
}

__device__ __forceinline__ void wait_async0() {
#if __has_builtin(__builtin_amdgcn_s_wait_asynccnt)
  __builtin_amdgcn_s_wait_asynccnt(0);
#else
  asm volatile("s_wait_asynccnt 0x0" ::: "memory");
#endif
}

// ---------------------------------------------------------------------------
// Prep: fp32 -> bf16 conversions / weight transposes (weights stored (N,K)
// row-major so each WMMA B-fragment lane reads two contiguous 16-byte runs).
// ---------------------------------------------------------------------------
__global__ __launch_bounds__(256) void k_cvt_x(const float* __restrict__ x,
                                               __bf16* __restrict__ xb, int n4) {
  int i = blockIdx.x * 256 + threadIdx.x;
  if (i < n4) {
    const float4 f = ((const float4*)x)[i];
    v4bf o; o[0] = (__bf16)f.x; o[1] = (__bf16)f.y; o[2] = (__bf16)f.z; o[3] = (__bf16)f.w;
    ((v4bf*)xb)[i] = o;
  }
}

// WprojT[h][e][d] = W_proj[h][d][e]   (output: 8 * 128 * 1024 bf16)
__global__ __launch_bounds__(256) void k_tr_wproj(const float* __restrict__ W,
                                                  __bf16* __restrict__ WT) {
  int idx = blockIdx.x * 256 + threadIdx.x;          // 1048576 total
  int d = idx & 1023, e = (idx >> 10) & 127, h = idx >> 17;
  WT[idx] = (__bf16)W[(size_t)h * (ND * NHD) + (size_t)d * NHD + e];
}

// WoutT[n][k] = W_out[k][n]          (output: 1024 * 1024 bf16)
__global__ __launch_bounds__(256) void k_tr_wout(const float* __restrict__ W,
                                                 __bf16* __restrict__ WT) {
  int idx = blockIdx.x * 256 + threadIdx.x;          // 1048576 total
  int k = idx & 1023, n = idx >> 10;
  WT[idx] = (__bf16)W[(size_t)k * ND + n];
}

// ---------------------------------------------------------------------------
// Phase 1: heads[h,b,s,e] = (x @ W_proj[h]) * cos(pi*freqs[h,e])
// Block: one (h, 16-row) tile; 8 waves cover the 8 N-tiles of HD=128.
// The shared 16x32 bf16 A-tile per K-step is staged in LDS with ping-pong
// async copies (waves 0-1 issue one b128 per lane), overlapped with WMMA.
//
// A-fragment (16-bit A 16x32, ISA 7.12.2): lane L<16 holds row L, K 0..7 in
// elems 0..7 and K 16..23 in elems 8..15; lane L>=16 holds K 8..15 / 24..31.
// B-fragment (32x16): same K order as A -- lane L<16 holds K 0..7 / 16..23 of
// column n, lane L>=16 holds K 8..15 / 24..31 (two 16B reads from WT[n][k]).
// ---------------------------------------------------------------------------
__global__ __launch_bounds__(256)
void k_heads_gemm(const __bf16* __restrict__ xb, const __bf16* __restrict__ WpT,
                  const float* __restrict__ freqs, float* __restrict__ heads,
                  float* __restrict__ hm) {
  __shared__ __bf16 sA[2][16][40];         // 32 K + 8 pad -> 80B rows (bank spread)
  const int tid  = threadIdx.x;
  const int wave = tid >> 5, lane = tid & 31;
  const int h    = blockIdx.x >> 11;       // / 2048 m-tiles
  const int mt   = blockIdx.x & 2047;
  const int row0 = mt << 4;                // global row in flattened (b,s)
  const int b    = row0 >> 13;             // / 8192
  const int s0   = row0 & 8191;
  const int n0   = wave << 4;              // 8 waves cover HD=128
  const int mr   = lane & 15;
  const int hi   = lane >> 4;              // lane half

  // staging map: threads 0..63 (waves 0,1), one 16B segment each per K-tile
  const int srow = tid >> 2, sseg = tid & 3;
  const __bf16* gsrc0 = xb + (size_t)(row0 + srow) * ND + sseg * 8;

  if (tid < 64) async_copy_b128_to_lds(gsrc0, &sA[0][srow][sseg * 8]);
  wait_async0();
  __syncthreads();

  const __bf16* bBase = WpT + (size_t)(h * NHD + n0 + mr) * ND;
  v8f acc = {};
#pragma unroll 2
  for (int kk = 0; kk < 32; ++kk) {
    const int buf = kk & 1;
    // prefetch next K-tile into the other buffer (overlaps with WMMA below)
    if (kk + 1 < 32 && tid < 64)
      async_copy_b128_to_lds(gsrc0 + (size_t)(kk + 1) * 32,
                             &sA[buf ^ 1][srow][sseg * 8]);

    const v8bf a0 = *(const v8bf*)(&sA[buf][mr][8 * hi]);
    const v8bf a1 = *(const v8bf*)(&sA[buf][mr][16 + 8 * hi]);
    const int k0 = kk << 5;
    const v8bf b0 = *(const v8bf*)(bBase + k0 + 8 * hi);
    const v8bf b1 = *(const v8bf*)(bBase + k0 + 8 * hi + 16);
    v16bf a, bm;
#pragma unroll
    for (int i = 0; i < 8; ++i) { a[i] = a0[i]; a[i + 8] = a1[i];
                                  bm[i] = b0[i]; bm[i + 8] = b1[i]; }
    acc = __builtin_amdgcn_wmma_f32_16x16x32_bf16(false, a, false, bm,
                                                  (short)0, acc, false, false);
    wait_async0();       // prefetch landed (issuing waves; no-op for others)
    __syncthreads();     // everyone done reading sA[buf] before it is reused
  }

  // column scale: cos(pi * freqs[h, n])
  const float cs = __cosf(PI_F * freqs[h * NHD + n0 + mr]);
  float csum = 0.f;
#pragma unroll
  for (int r = 0; r < 8; ++r) { acc[r] *= cs; csum += acc[r]; }

  // partial column sum over this lane's 8 rows -> sequence-mean accumulator
  atomicAdd(&hm[(h * NB + b) * NHD + n0 + mr], csum);

  // store the 16x16 tile: lane holds rows (r + 8*hi), column n0+mr
  float* hdst = heads + ((size_t)((h * NB + b) * NS + s0 + 8 * hi)) * NHD + n0 + mr;
#pragma unroll
  for (int r = 0; r < 8; ++r) hdst[(size_t)r * NHD] = acc[r];
}

// ---------------------------------------------------------------------------
// Phase 2 (tiny): impedance matrix + mixing coefficients, one block per batch.
// wave w <-> head h, lane <-> polarity p (wave32).
// ---------------------------------------------------------------------------
__global__ __launch_bounds__(256)
void k_impedance(const float* __restrict__ hm, const float* __restrict__ W_pol,
                 const float* __restrict__ b_pol, const float* __restrict__ W_imp1,
                 const float* __restrict__ b_imp1, const float* __restrict__ W_imp2,
                 const float* __restrict__ b_imp2, float* __restrict__ imp_out,
                 float* __restrict__ coeff) {
  __shared__ float sPm[NH][NP];
  const int b = blockIdx.x, t = threadIdx.x;
  const int h = t >> 5, p = t & 31;
  const float invS = 1.f / (float)NS;

  float dot = 0.f;
  for (int e = 0; e < NHD; ++e)
    dot += (hm[(h * NB + b) * NHD + e] * invS) * W_pol[(h * NHD + e) * NP + p];
  float pv = tanhf(dot + b_pol[h * NP + p]);

  float n2 = pv * pv;
  for (int off = 16; off; off >>= 1) n2 += __shfl_xor(n2, off, 32);
  sPm[h][p] = pv / fmaxf(sqrtf(n2), 1e-12f);
  __syncthreads();

  if (t < 64) {
    const int hh = t >> 3, g = t & 7;
    float dp = 0.f;
    for (int q = 0; q < NP; ++q) dp += sPm[hh][q] * sPm[g][q];
    float accum = b_imp2[0];
#pragma unroll
    for (int k = 0; k < 16; ++k) {
      float u  = dp * W_imp1[k] + b_imp1[k];
      float ge = 0.5f * u * (1.f + erff(u * 0.70710678118654752f));  // exact GELU
      accum += ge * W_imp2[k];
    }
    float imp = (accum > 20.f) ? accum : log1pf(expf(accum));        // softplus
    if (hh == g) imp = 0.f;
    imp_out[b * 64 + t] = imp;
    coeff[b * 64 + t]   = (hh == g) ? 0.f : 0.1f / (1.f + imp);
  }
}

// ---------------------------------------------------------------------------
// Phase 3: per 16-row tile -- head mixing into LDS (bf16 merged rows), then
// merged @ W_out with 16 waves x 4 column tiles of WMMA; writes
// proj + b_out + x (pre-LayerNorm) to y.
// LDS rows padded by 8 bf16 (16B) so lanes reading different rows hit
// different bank groups.
// ---------------------------------------------------------------------------
__global__ __launch_bounds__(512)
void k_mix_gemm(const float* __restrict__ heads, const float* __restrict__ coeff,
                const __bf16* __restrict__ WoT, const float* __restrict__ x,
                const float* __restrict__ b_out, float* __restrict__ y) {
  __shared__ __bf16 sM[16][ND + 8];
  __shared__ float  sC[64];
  const int t    = threadIdx.x;
  const int row0 = blockIdx.x << 4;
  const int b    = row0 >> 13, s0 = row0 & 8191;
  const float invS = 1.f / (float)NS;

  if (t < 64) sC[t] = coeff[b * 64 + t];
  __syncthreads();

  // mixing: out_heads[i] = heads[i] + ((s+1)/S) * sum_g coeff[i][g]*heads[g]
#pragma unroll
  for (int it = 0; it < 4; ++it) {
    const int q = t + (it << 9);        // 16 rows * 128 e = 2048 tasks
    const int m = q >> 7, e = q & 127;
    float hv[NH];
#pragma unroll
    for (int g = 0; g < NH; ++g)
      hv[g] = heads[((size_t)((g * NB + b) * NS + s0 + m)) * NHD + e];
    const float cscale = (float)(s0 + m + 1) * invS;
#pragma unroll
    for (int i = 0; i < NH; ++i) {
      float mix = 0.f;
#pragma unroll
      for (int g = 0; g < NH; ++g) mix += sC[i * 8 + g] * hv[g];
      sM[m][i * NHD + e] = (__bf16)(hv[i] + cscale * mix);
    }
  }
  __syncthreads();

  // GEMM: wave w owns output columns [w*64, w*64+64) as 4 WMMA tiles
  const int wave = t >> 5, lane = t & 31;
  const int mr = lane & 15, hi = lane >> 4;
  v8f acc[4] = {v8f{}, v8f{}, v8f{}, v8f{}};
  const __bf16* a_row = &sM[mr][0];

  for (int k0 = 0; k0 < ND; k0 += 32) {
    const v8bf a0 = *(const v8bf*)(a_row + k0 + 8 * hi);
    const v8bf a1 = *(const v8bf*)(a_row + k0 + 8 * hi + 16);
    v16bf a;
#pragma unroll
    for (int i = 0; i < 8; ++i) { a[i] = a0[i]; a[i + 8] = a1[i]; }
#pragma unroll
    for (int j = 0; j < 4; ++j) {
      const __bf16* brow = WoT + (size_t)(wave * 64 + j * 16 + mr) * ND + k0 + 8 * hi;
      const v8bf b0 = *(const v8bf*)(brow);
      const v8bf b1 = *(const v8bf*)(brow + 16);
      v16bf bm;
#pragma unroll
      for (int i = 0; i < 8; ++i) { bm[i] = b0[i]; bm[i + 8] = b1[i]; }
      acc[j] = __builtin_amdgcn_wmma_f32_16x16x32_bf16(false, a, false, bm,
                                                       (short)0, acc[j], false, false);
    }
  }

  // epilogue: y = proj + b_out + x   (LayerNorm done by next kernel, in place)
#pragma unroll
  for (int j = 0; j < 4; ++j) {
    const int c = wave * 64 + j * 16 + mr;
    const float bo = b_out[c];
#pragma unroll
    for (int r = 0; r < 8; ++r) {
      const int m = r + 8 * hi;
      const size_t oi = ((size_t)(b * NS + s0 + m)) * ND + c;
      y[oi] = acc[j][r] + bo + x[oi];
    }
  }
}

// ---------------------------------------------------------------------------
// Phase 4: in-place LayerNorm, one wave per row (32 lanes x 32 cols each).
// ---------------------------------------------------------------------------
__global__ __launch_bounds__(256)
void k_layernorm(float* __restrict__ y, const float* __restrict__ gamma,
                 const float* __restrict__ beta) {
  const int wave = threadIdx.x >> 5, lane = threadIdx.x & 31;
  const int row  = blockIdx.x * 8 + wave;
  float* p = y + (size_t)row * ND;
  float v[32], s = 0.f, ss = 0.f;
#pragma unroll
  for (int i = 0; i < 32; ++i) {
    const float u = p[i * 32 + lane];
    v[i] = u; s += u; ss += u * u;
  }
  for (int off = 16; off; off >>= 1) {
    s  += __shfl_xor(s,  off, 32);
    ss += __shfl_xor(ss, off, 32);
  }
  const float mean = s * (1.f / ND);
  const float var  = ss * (1.f / ND) - mean * mean;
  const float rstd = rsqrtf(var + 1e-5f);
#pragma unroll
  for (int i = 0; i < 32; ++i) {
    const int c = i * 32 + lane;
    p[c] = (v[i] - mean) * rstd * gamma[c] + beta[c];
  }
}

// ---------------------------------------------------------------------------
// Workspace layout (bytes):
//   [0,             67108864)  xb     bf16 [B*S*D]
//   [67108864,      69206016)  WprojT bf16 [8][128][1024]
//   [69206016,      71303168)  WoutT  bf16 [1024][1024]   (n,k)
//   [71303168,     205520896)  heads  f32  [8][4][8192][128]
//   [205520896,    205537280)  hm     f32  [8][4][128]
//   [205537280,    205538304)  coeff  f32  [4][8][8]
// total ~196 MB.
// ---------------------------------------------------------------------------
extern "C" void kernel_launch(void* const* d_in, const int* in_sizes, int n_in,
                              void* d_out, int out_size, void* d_ws, size_t ws_size,
                              hipStream_t stream) {
  const float* x      = (const float*)d_in[0];
  const float* W_proj = (const float*)d_in[1];
  const float* freqs  = (const float*)d_in[2];
  const float* W_pol  = (const float*)d_in[3];
  const float* b_pol  = (const float*)d_in[4];
  const float* W_imp1 = (const float*)d_in[5];
  const float* b_imp1 = (const float*)d_in[6];
  const float* W_imp2 = (const float*)d_in[7];
  const float* b_imp2 = (const float*)d_in[8];
  const float* W_out  = (const float*)d_in[9];
  const float* b_out  = (const float*)d_in[10];
  const float* gamma  = (const float*)d_in[11];
  const float* beta   = (const float*)d_in[12];

  char* ws = (char*)d_ws;
  __bf16* xb    = (__bf16*)(ws);
  __bf16* WpT   = (__bf16*)(ws + 67108864);
  __bf16* WoT   = (__bf16*)(ws + 69206016);
  float*  heads = (float*)(ws + 71303168);
  float*  hm    = (float*)(ws + 205520896);
  float*  coeff = (float*)(ws + 205537280);

  float* y       = (float*)d_out;             // [B,S,D]
  float* imp_out = y + (size_t)NB * NS * ND;  // [B,8,8] appended

  hipMemsetAsync(hm, 0, NH * NB * NHD * sizeof(float), stream);  // graph-safe

  k_cvt_x   <<<32768, 256, 0, stream>>>(x, xb, (NB * NS * ND) / 4);
  k_tr_wproj<<< 4096, 256, 0, stream>>>(W_proj, WpT);
  k_tr_wout <<< 4096, 256, 0, stream>>>(W_out, WoT);

  k_heads_gemm<<<NH * (NM / 16), 256, 0, stream>>>(xb, WpT, freqs, heads, hm);
  k_impedance <<<NB, 256, 0, stream>>>(hm, W_pol, b_pol, W_imp1, b_imp1,
                                       W_imp2, b_imp2, imp_out, coeff);
  k_mix_gemm  <<<NM / 16, 512, 0, stream>>>(heads, coeff, WoT, x, b_out, y);
  k_layernorm <<<NM / 8, 256, 0, stream>>>(y, gamma, beta);
}
